// MultiHeadAttention_37349035606780
// MI455X (gfx1250) — matrix operand order repaired and checked by reading, not compiled, not run on hardware
//
#include <hip/hip_runtime.h>
#include <hip/hip_bf16.h>
#include <math.h>
#include <stdint.h>

typedef _Float16 f16;
typedef __attribute__((ext_vector_type(16))) _Float16 v16h;
typedef __attribute__((ext_vector_type(8)))  _Float16 v8h;
typedef __attribute__((ext_vector_type(8)))  float    v8f;

#define N_TOK 4096
#define D_MODEL 256
#define HEADS 8
#define HEAD_DIM 32

__device__ __forceinline__ v8f wmma_f16(v16h a, v16h b, v8f c) {
  return __builtin_amdgcn_wmma_f32_16x16x32_f16(false, a, false, b, (short)0, c,
                                                false, false);
}

// ---------------------------------------------------------------------------
// Cast x -> f16 row-major; cast W -> f16 TRANSPOSED (WT[col][row]) so that
// WMMA B-fragments are contiguous 32-byte loads per lane.
// ---------------------------------------------------------------------------
__global__ __launch_bounds__(256) void cast_kernel(
    const float* __restrict__ x,
    const float* __restrict__ wq, const float* __restrict__ wk,
    const float* __restrict__ wv, const float* __restrict__ wo,
    f16* __restrict__ x16,
    f16* __restrict__ wqT, f16* __restrict__ wkT,
    f16* __restrict__ wvT, f16* __restrict__ woT)
{
  int i = blockIdx.x * blockDim.x + threadIdx.x;   // 0 .. 4096*256-1
  x16[i] = (f16)x[i];
  if (i < D_MODEL * D_MODEL) {
    int r = i >> 8, c = i & 255;
    wqT[c * D_MODEL + r] = (f16)wq[i];
    wkT[c * D_MODEL + r] = (f16)wk[i];
    wvT[c * D_MODEL + r] = (f16)wv[i];
    woT[c * D_MODEL + r] = (f16)wo[i];
  }
}

// ---------------------------------------------------------------------------
// C[M x 256] = A16[M x 256] @ (BT16)^T + bias    (BT16 is [256 x 256], BT[n][k])
// mode 0: f16 row-major   out16[row*256+col]
// mode 1: f16 transposed  out16[col*M+row]    (for V^T)
// mode 2: f32 row-major   outf [row*256+col]  (final output)
// One wave computes a 32(M) x 64(N) tile: each B fragment feeds 2 WMMAs.
// ---------------------------------------------------------------------------
__global__ __launch_bounds__(256) void gemm_wmma(
    const f16* __restrict__ A, const f16* __restrict__ BT,
    const float* __restrict__ bias, f16* __restrict__ out16,
    float* __restrict__ outf, int M, int mode)
{
  int wid  = blockIdx.x * (blockDim.x >> 5) + (threadIdx.x >> 5);
  int lane = threadIdx.x & 31;
  int tm = wid >> 2;           // 32-row M-tile: 0..M/32-1
  int tg = wid & 3;            // 64-col N-group: 0..3
  int m  = lane & 15;
  int kh = (lane >> 4) & 1;

  v8f zero = {};
  v8f acc[2][4];
  #pragma unroll
  for (int h = 0; h < 2; ++h)
    #pragma unroll
    for (int t = 0; t < 4; ++t) acc[h][t] = zero;

  const f16* arow0 = A + (size_t)(tm * 32 + m) * D_MODEL;
  const f16* arow1 = arow0 + 16 * D_MODEL;

  for (int kb = 0; kb < 8; ++kb) {
    v16h af0, af1;
    {
      v8h a0 = *(const v8h*)(arow0 + kb * 32 + 16 * kh);
      v8h a1 = *(const v8h*)(arow0 + kb * 32 + 8 + 16 * kh);
      v8h a2 = *(const v8h*)(arow1 + kb * 32 + 16 * kh);
      v8h a3 = *(const v8h*)(arow1 + kb * 32 + 8 + 16 * kh);
      #pragma unroll
      for (int i = 0; i < 8; ++i) {
        af0[i] = a0[i]; af0[8 + i] = a1[i];
        af1[i] = a2[i]; af1[8 + i] = a3[i];
      }
    }
    #pragma unroll
    for (int t = 0; t < 4; ++t) {
      int col = tg * 64 + t * 16 + m;    // n = lane&15
      v16h bf = *(const v16h*)(BT + (size_t)col * D_MODEL + kb * 32 + 16 * kh);
      acc[0][t] = wmma_f16(af0, bf, acc[0][t]);
      acc[1][t] = wmma_f16(af1, bf, acc[1][t]);
    }
  }

  #pragma unroll
  for (int h = 0; h < 2; ++h) {
    #pragma unroll
    for (int t = 0; t < 4; ++t) {
      int col = tg * 64 + t * 16 + m;
      float bv = bias[col];
      #pragma unroll
      for (int r = 0; r < 8; ++r) {
        int row = tm * 32 + h * 16 + r + 8 * kh;
        float v = acc[h][t][r] + bv;
        if (mode == 0)      out16[(size_t)row * D_MODEL + col] = (f16)v;
        else if (mode == 1) out16[(size_t)col * M + row]       = (f16)v;
        else                outf [(size_t)row * D_MODEL + col] = v;
      }
    }
  }
}

// ---------------------------------------------------------------------------
// Async adj tile staging: 256 threads x 8 bytes -> 16x32 fp32 LDS tile,
// directly to LDS via the CDNA5 async path (ASYNCcnt), no VGPR round-trip.
// ---------------------------------------------------------------------------
__device__ __forceinline__ void stage_adj_async(const float* __restrict__ adj,
                                                int q0, int k0,
                                                float (*dst)[32]) {
  int qi = threadIdx.x >> 4;             // 0..15 query row
  int c0 = (threadIdx.x & 15) * 2;       // 0..30 key col pair
  unsigned lds_off = (unsigned)(uintptr_t)&dst[qi][c0];
  unsigned voff    = (unsigned)((qi * N_TOK + c0) * 4);
  const float* base = adj + (size_t)q0 * N_TOK + k0;
  asm volatile("global_load_async_to_lds_b64 %0, %1, %2"
               :
               : "v"(lds_off), "v"(voff), "s"(base)
               : "memory");
}

// ---------------------------------------------------------------------------
// Flash attention: block = 16 query rows x 8 heads (8 waves, 1 head each).
// Double-buffered async adj staging, one barrier per 32-key block.
// Per block: 2 WMMAs (S = Q K^T), fp32 online softmax, P transpose via
// per-wave LDS, 2 WMMAs (O += P V). Output O16 f16 row-major [N][256].
// ---------------------------------------------------------------------------
__global__ __launch_bounds__(256) void flash_attn(
    const f16* __restrict__ Q16, const f16* __restrict__ K16,
    const f16* __restrict__ Vt16, const float* __restrict__ adj,
    f16* __restrict__ O16)
{
  __shared__ float s_adj[2][16][32];
  __shared__ f16   s_p[HEADS][16][32];

  const int q0   = blockIdx.x * 16;
  const int h    = threadIdx.x >> 5;     // head = wave id
  const int lane = threadIdx.x & 31;
  const int m    = lane & 15;
  const int kh   = (lane >> 4) & 1;
  const float scale = 0.1767766952966369f;   // 1/sqrt(32)
  const float NINF = -__builtin_inff();

  // Q fragment (A-layout), loaded once per wave
  v16h qf;
  {
    const f16* qrow = Q16 + (size_t)(q0 + m) * D_MODEL + h * HEAD_DIM;
    v8h a0 = *(const v8h*)(qrow + 16 * kh);
    v8h a1 = *(const v8h*)(qrow + 8 + 16 * kh);
    #pragma unroll
    for (int i = 0; i < 8; ++i) { qf[i] = a0[i]; qf[8 + i] = a1[i]; }
  }

  v8f o0 = {}, o1 = {};
  float mrow[8], lrow[8];
  #pragma unroll
  for (int r = 0; r < 8; ++r) { mrow[r] = NINF; lrow[r] = 0.f; }

  // prologue: stage first adj tile
  stage_adj_async(adj, q0, 0, s_adj[0]);

  for (int k0 = 0; k0 < N_TOK; k0 += 32) {
    const int buf = (k0 >> 5) & 1;

    // current tile landed; barrier also proves nobody still reads buf^1
    asm volatile("s_wait_asynccnt 0x0" ::: "memory");
    __syncthreads();
    if (k0 + 32 < N_TOK) stage_adj_async(adj, q0, k0 + 32, s_adj[buf ^ 1]);

    // ---- S = Q K^T  (two 16x16 WMMAs covering 32 keys) ----
    v16h b0 = *(const v16h*)(K16 + (size_t)(k0 + m) * D_MODEL +
                             h * HEAD_DIM + 16 * kh);
    v16h b1 = *(const v16h*)(K16 + (size_t)(k0 + 16 + m) * D_MODEL +
                             h * HEAD_DIM + 16 * kh);
    v8f zero = {};
    v8f s0 = wmma_f16(qf, b0, zero);
    v8f s1 = wmma_f16(qf, b1, zero);

    // ---- mask + online softmax (rows r+8*kh, col n = m / 16+m) ----
    #pragma unroll
    for (int r = 0; r < 8; ++r) {
      int row = r + 8 * kh;
      float v0 = (s_adj[buf][row][m]      > 0.f) ? s0[r] * scale : NINF;
      float v1 = (s_adj[buf][row][16 + m] > 0.f) ? s1[r] * scale : NINF;
      float mb = fmaxf(v0, v1);
      for (int d = 1; d < 16; d <<= 1) mb = fmaxf(mb, __shfl_xor(mb, d, 32));
      float mnew  = fmaxf(mrow[r], mb);
      float alpha = (mnew == mrow[r]) ? 1.f : __expf(mrow[r] - mnew);
      float p0 = (v0 == NINF) ? 0.f : __expf(v0 - mnew);
      float p1 = (v1 == NINF) ? 0.f : __expf(v1 - mnew);
      float ps = p0 + p1;
      for (int d = 1; d < 16; d <<= 1) ps += __shfl_xor(ps, d, 32);
      lrow[r] = lrow[r] * alpha + ps;
      mrow[r] = mnew;
      o0[r] *= alpha; o1[r] *= alpha;
      s0[r] = p0; s1[r] = p1;
    }

    // ---- transpose P: C-layout -> A-layout through per-wave LDS ----
    #pragma unroll
    for (int r = 0; r < 8; ++r) {
      int row = r + 8 * kh;
      s_p[h][row][m]      = (f16)s0[r];
      s_p[h][row][16 + m] = (f16)s1[r];
    }
    v16h pf;   // same-wave LDS RAW: DS ops are in-order per wave
    {
      const f16* pr = &s_p[h][m][0];
      v8h a0 = *(const v8h*)(pr + 16 * kh);
      v8h a1 = *(const v8h*)(pr + 8 + 16 * kh);
      #pragma unroll
      for (int i = 0; i < 8; ++i) { pf[i] = a0[i]; pf[8 + i] = a1[i]; }
    }

    // ---- O += P V  (V stored transposed: Vt[hd][token]) ----
    v16h vb0 = *(const v16h*)(Vt16 + (size_t)(h * HEAD_DIM + m) * N_TOK +
                              k0 + 16 * kh);
    v16h vb1 = *(const v16h*)(Vt16 + (size_t)(h * HEAD_DIM + 16 + m) * N_TOK +
                              k0 + 16 * kh);
    o0 = wmma_f16(pf, vb0, o0);
    o1 = wmma_f16(pf, vb1, o1);
  }

  // ---- normalize and store O (f16 row-major) ----
  #pragma unroll
  for (int r = 0; r < 8; ++r) {
    float inv = (lrow[r] > 0.f) ? 1.f / lrow[r] : 0.f;
    int row = q0 + r + 8 * kh;
    O16[(size_t)row * D_MODEL + h * HEAD_DIM + m]      = (f16)(o0[r] * inv);
    O16[(size_t)row * D_MODEL + h * HEAD_DIM + 16 + m] = (f16)(o1[r] * inv);
  }
}

// ---------------------------------------------------------------------------
extern "C" void kernel_launch(void* const* d_in, const int* in_sizes, int n_in,
                              void* d_out, int out_size, void* d_ws, size_t ws_size,
                              hipStream_t stream) {
  (void)in_sizes; (void)n_in; (void)out_size; (void)ws_size;
  const float* x   = (const float*)d_in[0];
  const float* adj = (const float*)d_in[1];
  const float* Wq  = (const float*)d_in[2];
  const float* bq  = (const float*)d_in[3];
  const float* Wk  = (const float*)d_in[4];
  const float* bk  = (const float*)d_in[5];
  const float* Wv  = (const float*)d_in[6];
  const float* bv  = (const float*)d_in[7];
  const float* Wo  = (const float*)d_in[8];
  const float* bo  = (const float*)d_in[9];
  float* out = (float*)d_out;

  char* ws = (char*)d_ws;
  const size_t MB = 1u << 20;
  f16* X16  = (f16*)(ws + 0 * MB);        // 2 MB
  f16* Q16  = (f16*)(ws + 2 * MB);        // 2 MB
  f16* K16  = (f16*)(ws + 4 * MB);        // 2 MB
  f16* Vt16 = (f16*)(ws + 6 * MB);        // 2 MB  [256][4096]
  f16* O16  = (f16*)(ws + 8 * MB);        // 2 MB
  f16* WqT  = (f16*)(ws + 10 * MB);       // 128 KB each
  f16* WkT  = (f16*)(ws + 10 * MB + 128 * 1024);
  f16* WvT  = (f16*)(ws + 10 * MB + 256 * 1024);
  f16* WoT  = (f16*)(ws + 10 * MB + 384 * 1024);

  cast_kernel<<<(N_TOK * D_MODEL) / 256, 256, 0, stream>>>(
      x, Wq, Wk, Wv, Wo, X16, WqT, WkT, WvT, WoT);

  // 512 wave-tiles (128 M-tiles of 32 rows x 4 N-groups), 8 waves per block
  gemm_wmma<<<64, 256, 0, stream>>>(X16, WqT, bq, Q16, nullptr, N_TOK, 0);
  gemm_wmma<<<64, 256, 0, stream>>>(X16, WkT, bk, K16, nullptr, N_TOK, 0);
  gemm_wmma<<<64, 256, 0, stream>>>(X16, WvT, bv, Vt16, nullptr, N_TOK, 1);

  flash_attn<<<N_TOK / 16, 256, 0, stream>>>(Q16, K16, Vt16, adj, O16);

  gemm_wmma<<<64, 256, 0, stream>>>(O16, WoT, bo, nullptr, out, N_TOK, 2);
}
